// Attention_37958920962076
// MI455X (gfx1250) — compile-verified, matrix operand order checked by reading, not run on hardware
//
#include <hip/hip_runtime.h>
#include <hip/hip_bf16.h>

// ---------------- problem constants ----------------
constexpr int BATCH = 4;
constexpr int SEQ   = 2048;
constexpr int DIM   = 1024;
constexpr int HEADS = 16;
constexpr int DH    = 64;
constexpr int ROWS  = BATCH * SEQ;     // 8192
constexpr int NKPAD = 2080;            // 2049 keys padded to multiple of 32
constexpr float QSCALE = 0.125f;       // 64^-0.5

// ---------------- workspace layout (bytes) ----------------
constexpr size_t XN_OFF    = 0;                                          // xn bf16, reused as attn-out bf16
constexpr size_t WQT_OFF   = (size_t)ROWS * DIM * 2;                     // 16 MiB
constexpr size_t WKVT_OFF  = WQT_OFF  + (size_t)DIM * DIM * 2;           // +2 MiB
constexpr size_t WOUTT_OFF = WKVT_OFF + (size_t)DIM * 2048 * 2;          // +4 MiB
constexpr size_t Q_OFF     = WOUTT_OFF + (size_t)DIM * DIM * 2;          // +2 MiB
constexpr size_t K_OFF     = Q_OFF + (size_t)BATCH * HEADS * SEQ * DH * 2;      // +16 MiB
constexpr size_t VT_OFF    = K_OFF + (size_t)BATCH * HEADS * NKPAD * DH * 2;    // +16.25 MiB

// ---------------- types ----------------
typedef __attribute__((ext_vector_type(16))) __bf16 v16bf;
typedef __attribute__((ext_vector_type(8)))  float  v8f;
typedef int v4i __attribute__((__vector_size__(4 * sizeof(int))));
typedef unsigned short ush;

union FragU { uint4 u[2]; v16bf v; };

// ---------------- helpers ----------------
__device__ __forceinline__ ush f2bf(float f) {
  unsigned u = __builtin_bit_cast(unsigned, f);
  u += 0x7FFFu + ((u >> 16) & 1u);               // round-nearest-even
  return (ush)(u >> 16);
}

template <int IMM>
__device__ __forceinline__ float swz(float x) {   // ds_swizzle_b32 (imm pattern)
  return __builtin_bit_cast(float,
      __builtin_amdgcn_ds_swizzle(__builtin_bit_cast(int, x), IMM));
}
__device__ __forceinline__ float bcast(float x, int srcLane) { // ds_bpermute_b32
  return __builtin_bit_cast(float,
      __builtin_amdgcn_ds_bpermute(srcLane << 2, __builtin_bit_cast(int, x)));
}
__device__ __forceinline__ float wave_sum(float x) {
  x += swz<0x401F>(x); x += swz<0x201F>(x); x += swz<0x101F>(x);
  x += swz<0x081F>(x); x += swz<0x041F>(x);
  return x;
}

// 16-byte async global->LDS copy (gfx1250 ASYNCcnt path), with sync fallback.
__device__ __forceinline__ void cp_async16(void* lds, const void* g) {
#if __has_builtin(__builtin_amdgcn_global_load_async_to_lds_b128)
  __builtin_amdgcn_global_load_async_to_lds_b128(
      (__attribute__((address_space(1))) v4i*)(g),
      (__attribute__((address_space(3))) v4i*)(lds), 0, 0);
#else
  *(uint4*)lds = *(const uint4*)g;
#endif
}
template <int N>
__device__ __forceinline__ void cp_wait() {
#if __has_builtin(__builtin_amdgcn_global_load_async_to_lds_b128)
# if __has_builtin(__builtin_amdgcn_s_wait_asynccnt)
  __builtin_amdgcn_s_wait_asynccnt(N);
# else
  asm volatile("s_wait_asynccnt %0" :: "i"(N) : "memory");
# endif
#endif
}

// Fragment load: CDNA5 16-bit A-layout == B "column-major" layout.
// lane 0-15: row = base+lane,   K = kbase+{0..7} and kbase+16+{0..7}
// lane16-31: row = base+lane-16,K = kbase+8+{0..7} and kbase+24+{0..7}
// Works for global (global_load_b128) and LDS (ds_load_b128) pointers.
__device__ __forceinline__ v16bf load_frag(const ush* base, int ld, int row,
                                           int kbase, int lane) {
  const int half = lane >> 4;
  const ush* p = base + (size_t)(row + (lane & 15)) * ld + kbase + half * 8;
  FragU f;
  f.u[0] = *(const uint4*)(p);
  f.u[1] = *(const uint4*)(p + 16);
  return f.v;
}

__device__ __forceinline__ v8f wmma_bf16(v16bf a, v16bf b, v8f c) {
  return __builtin_amdgcn_wmma_f32_16x16x32_bf16(false, a, false, b,
                                                 (short)0, c, false, false);
}

// ---------------- kernel: cast + transpose weights f32 (KxN) -> bf16 (NxK) ----------------
__global__ void __launch_bounds__(256) castT_kernel(const float* __restrict__ src,
                                                    ush* __restrict__ dst, int K, int N) {
  int idx = blockIdx.x * 256 + threadIdx.x;
  if (idx >= K * N) return;
  int n = idx / K, k = idx - n * K;
  dst[idx] = f2bf(src[(size_t)k * N + n]);
}

// ---------------- kernel: null-kv + pad init ----------------
__global__ void init_kv_kernel(const float* __restrict__ nullkv,
                               ush* __restrict__ Kd, ush* __restrict__ Vt) {
  int bh = blockIdx.x;          // 64
  int h  = bh & 15;
  int d  = threadIdx.x;         // 64
  Kd[(size_t)bh * NKPAD * DH + d]   = f2bf(nullkv[h * DH + d]);          // key 0
  Vt[((size_t)bh * DH + d) * NKPAD] = f2bf(nullkv[(HEADS + h) * DH + d]);
  for (int j = SEQ + 1; j < NKPAD; ++j) {
    Kd[((size_t)bh * NKPAD + j) * DH + d] = 0;
    Vt[((size_t)bh * DH + d) * NKPAD + j] = 0;
  }
}

// ---------------- kernel: LayerNorm f32 -> bf16 ----------------
__global__ void __launch_bounds__(256) ln_kernel(const float* __restrict__ x,
                                                 const float* __restrict__ gamma,
                                                 ush* __restrict__ xn) {
  const int row = blockIdx.x, t = threadIdx.x;
  const float4 v = ((const float4*)(x + (size_t)row * DIM))[t];
  __shared__ float red[8];

  float s = wave_sum(v.x + v.y + v.z + v.w);
  if ((t & 31) == 0) red[t >> 5] = s;
  __syncthreads();
  float tot = 0.f;
#pragma unroll
  for (int i = 0; i < 8; ++i) tot += red[i];
  const float mu = tot * (1.f / DIM);

  const float d0 = v.x - mu, d1 = v.y - mu, d2 = v.z - mu, d3 = v.w - mu;
  float sq = wave_sum(d0 * d0 + d1 * d1 + d2 * d2 + d3 * d3);
  __syncthreads();
  if ((t & 31) == 0) red[t >> 5] = sq;
  __syncthreads();
  float tv = 0.f;
#pragma unroll
  for (int i = 0; i < 8; ++i) tv += red[i];
  const float rs = rsqrtf(tv * (1.f / DIM) + 1e-5f);

  const float4 g = ((const float4*)gamma)[t];
  const size_t o = (size_t)row * DIM + t * 4;
  xn[o + 0] = f2bf(d0 * rs * g.x);
  xn[o + 1] = f2bf(d1 * rs * g.y);
  xn[o + 2] = f2bf(d2 * rs * g.z);
  xn[o + 3] = f2bf(d3 * rs * g.w);
}

// ---------------- staged GEMM core: C[256x64] per block, 32x64 per wave, K=1024 ----------------
// LDS strides padded for conflict-free ds_load_b128 (80B rows, 16B aligned).
constexpr int ASTR = 40;   // 32 data + 8 pad (ush)
constexpr int BSTR = 40;
constexpr int LA_SZ = 256 * ASTR;   // per buffer (20 KiB)
constexpr int LB_SZ = 64 * BSTR;    // per buffer (5 KiB)

__device__ __forceinline__ void gemm_core(const ush* __restrict__ A,
                                          const ush* __restrict__ Bt,
                                          int m0blk, int n0blk,
                                          int wave, int lane,
                                          ush* lA, ush* lB, v8f acc[2][4]) {
  const int t = wave * 32 + lane;
  const int ar = t >> 2, ac = t & 3;      // 64 row-groups x 4 x 16B units
  auto issue = [&](int buf, int k) {
    ush* la = lA + buf * LA_SZ;
    ush* lb = lB + buf * LB_SZ;
#pragma unroll
    for (int q = 0; q < 4; ++q)           // A: 256 rows
      cp_async16(la + (ar + q * 64) * ASTR + ac * 8,
                 A + (size_t)(m0blk + ar + q * 64) * DIM + k + ac * 8);
    cp_async16(lb + ar * BSTR + ac * 8,   // B: 64 rows
               Bt + (size_t)(n0blk + ar) * DIM + k + ac * 8);
  };
  issue(0, 0);
  constexpr int KS = DIM / 32;
  // unroll 1: keep one iteration's fragments live (spill-free); overlap comes
  // from the async double-buffer, not from software unrolling.
#pragma unroll 1
  for (int i = 0; i < KS; ++i) {
    if (i + 1 < KS) { issue((i + 1) & 1, (i + 1) * 32); cp_wait<5>(); }
    else            { cp_wait<0>(); }
    __syncthreads();
    const ush* la = lA + (i & 1) * LA_SZ;
    const ush* lb = lB + (i & 1) * LB_SZ;
    v16bf a0 = load_frag(la, ASTR, wave * 32,      0, lane);
    v16bf a1 = load_frag(la, ASTR, wave * 32 + 16, 0, lane);
#pragma unroll
    for (int j = 0; j < 4; ++j) {
      v16bf b = load_frag(lb, BSTR, j * 16, 0, lane);
      acc[0][j] = wmma_bf16(a0, b, acc[0][j]);
      acc[1][j] = wmma_bf16(a1, b, acc[1][j]);
    }
    __syncthreads();
  }
}

// ---------------- kernel: Q projection (scaled, head-split store) ----------------
__global__ void __launch_bounds__(256) gemm_q_kernel(const ush* __restrict__ xn,
                                                     const ush* __restrict__ wqt,
                                                     ush* __restrict__ qout) {
  __shared__ ush lA[2 * LA_SZ];
  __shared__ ush lB[2 * LB_SZ];
  const int lane = threadIdx.x & 31, wave = threadIdx.x >> 5;
  const int m0 = blockIdx.x * 256, n0 = blockIdx.y * 64;
  v8f acc[2][4] = {};
  gemm_core(xn, wqt, m0, n0, wave, lane, lA, lB, acc);
  const int half = lane >> 4, nsub = lane & 15;
#pragma unroll
  for (int s = 0; s < 2; ++s)
#pragma unroll
    for (int j = 0; j < 4; ++j) {
      const int c = n0 + j * 16 + nsub, h = c >> 6, d = c & 63;
#pragma unroll
      for (int r = 0; r < 8; ++r) {
        const int R = m0 + wave * 32 + s * 16 + half * 8 + r, b = R >> 11, n = R & 2047;
        qout[((size_t)((b * HEADS + h) * SEQ + n)) * DH + d] = f2bf(acc[s][j][r] * QSCALE);
      }
    }
}

// ---------------- kernel: KV projection (K head-split, V transposed) ----------------
__global__ void __launch_bounds__(256) gemm_kv_kernel(const ush* __restrict__ xn,
                                                      const ush* __restrict__ wkvt,
                                                      ush* __restrict__ kout,
                                                      ush* __restrict__ vt) {
  __shared__ ush lA[2 * LA_SZ];
  __shared__ ush lB[2 * LB_SZ];
  const int lane = threadIdx.x & 31, wave = threadIdx.x >> 5;
  const int m0 = blockIdx.x * 256, n0 = blockIdx.y * 64;
  v8f acc[2][4] = {};
  gemm_core(xn, wkvt, m0, n0, wave, lane, lA, lB, acc);
  const int half = lane >> 4, nsub = lane & 15;
#pragma unroll
  for (int s = 0; s < 2; ++s)
#pragma unroll
    for (int j = 0; j < 4; ++j) {
      const int c = n0 + j * 16 + nsub;
#pragma unroll
      for (int r = 0; r < 8; ++r) {
        const int R = m0 + wave * 32 + s * 16 + half * 8 + r, b = R >> 11, n = R & 2047;
        const ush bv = f2bf(acc[s][j][r]);
        if (c < DIM) {                     // K half
          const int h = c >> 6, d = c & 63;
          kout[((size_t)(b * HEADS + h) * NKPAD + (n + 1)) * DH + d] = bv;
        } else {                           // V half, transposed store
          const int cc = c - DIM, h = cc >> 6, d = cc & 63;
          vt[((size_t)(b * HEADS + h) * DH + d) * NKPAD + (n + 1)] = bv;
        }
      }
    }
}

// ---------------- kernel: flash attention, block-shared async K/V staging ----------------
constexpr int KSTR = 72;   // 64 data + 8 pad (ush), 144B rows (16B aligned)
constexpr int VSTR = 40;   // 32 data + 8 pad
constexpr int LK_SZ = 32 * KSTR;
constexpr int LV_SZ = 64 * VSTR;

__global__ void __launch_bounds__(256) attn_kernel(const ush* __restrict__ Q,
                                                   const ush* __restrict__ K,
                                                   const ush* __restrict__ Vt,
                                                   const unsigned char* __restrict__ cmask,
                                                   ush* __restrict__ attn) {
  __shared__ ush lK[2 * LK_SZ];
  __shared__ ush lV[2 * LV_SZ];
  __shared__ float maskadd[NKPAD];

  const int bh = blockIdx.x >> 4;
  const int b = bh >> 4, h = bh & 15;
  const int qblk = (blockIdx.x & 15) * 128;
  const int t = threadIdx.x;
  for (int j = t; j < NKPAD; j += 256) {
    const bool ok = (j == 0) || (j <= SEQ && cmask[(size_t)b * SEQ + (j - 1)] != 0);
    maskadd[j] = ok ? 0.f : -1e30f;
  }

  const int lane = t & 31, wave = t >> 5;
  const int half = lane >> 4, nsub = lane & 15;
  const int qbase = qblk + wave * 16;
  const ush* Qb = Q  + (size_t)bh * SEQ * DH;
  const ush* Kb = K  + (size_t)bh * NKPAD * DH;
  const ush* Vb = Vt + (size_t)bh * DH * NKPAD;

  const v16bf qlo = load_frag(Qb, DH, qbase, 0, lane);
  const v16bf qhi = load_frag(Qb, DH, qbase, 32, lane);

  // per-chunk cooperative copy coords
  const int kr = t >> 3, kc8 = t & 7;     // K chunk: 32 rows x 8 x 16B
  const int vr = t >> 2, vc8 = t & 3;     // V chunk: 64 rows x 4 x 16B
  auto issueKV = [&](int buf, int kc) {
    cp_async16(&lK[buf * LK_SZ + kr * KSTR + kc8 * 8], Kb + (size_t)(kc + kr) * DH + kc8 * 8);
    cp_async16(&lV[buf * LV_SZ + vr * VSTR + vc8 * 8], Vb + (size_t)vr * NKPAD + kc + vc8 * 8);
  };

  v8f o[4] = {};
  float m = -1e30f, ssum = 0.f;

  issueKV(0, 0);
  constexpr int CHUNKS = NKPAD / 32;      // 65
#pragma unroll 1
  for (int i = 0; i < CHUNKS; ++i) {
    const int kc = i * 32;
    if (i + 1 < CHUNKS) { issueKV((i + 1) & 1, kc + 32); cp_wait<2>(); }
    else                { cp_wait<0>(); }
    __syncthreads();
    const ush* lk = &lK[(i & 1) * LK_SZ];
    const ush* lv = &lV[(i & 1) * LV_SZ];

    // S^T = K_tile x Q^T (two 16-key tiles, K-dim = 64 -> 2 wmma each)
    v16bf a0lo = load_frag(lk, KSTR, 0,  0,  lane);
    v16bf a0hi = load_frag(lk, KSTR, 0,  32, lane);
    v16bf a1lo = load_frag(lk, KSTR, 16, 0,  lane);
    v16bf a1hi = load_frag(lk, KSTR, 16, 32, lane);
    v8f s0 = {}; s0 = wmma_bf16(a0lo, qlo, s0); s0 = wmma_bf16(a0hi, qhi, s0);
    v8f s1 = {}; s1 = wmma_bf16(a1lo, qlo, s1); s1 = wmma_bf16(a1hi, qhi, s1);

    float p0[8], p1[8], lm = -1e30f;
#pragma unroll
    for (int r = 0; r < 8; ++r) {
      const float v0 = s0[r] + maskadd[kc + half * 8 + r];
      const float v1 = s1[r] + maskadd[kc + 16 + half * 8 + r];
      p0[r] = v0; p1[r] = v1;
      lm = fmaxf(lm, fmaxf(v0, v1));
    }
    lm = fmaxf(lm, swz<0x401F>(lm));           // combine lane halves (same query)
    const float mn = fmaxf(m, lm);
    const float scale = __expf(m - mn);
    float ls = 0.f;
#pragma unroll
    for (int r = 0; r < 8; ++r) {
      p0[r] = __expf(p0[r] - mn);
      p1[r] = __expf(p1[r] - mn);
      ls += p0[r] + p1[r];
    }
    ls += swz<0x401F>(ls);
    ssum = ssum * scale + ls;
    m = mn;

    // pack P^T chunk (32-key contraction) into bf16 A-fragment — no cross-lane moves
    union { v16bf v; ush s[16]; } pf;
#pragma unroll
    for (int r = 0; r < 8; ++r) { pf.s[r] = f2bf(p0[r]); pf.s[8 + r] = f2bf(p1[r]); }

    // rescale running output: per-query scale broadcast via ds_bpermute
#pragma unroll
    for (int r = 0; r < 8; ++r) {
      const float sc = bcast(scale, half * 8 + r);
      o[0][r] *= sc; o[1][r] *= sc; o[2][r] *= sc; o[3][r] *= sc;
    }

    // O += P x V (B-fragments from LDS-staged transposed V)
#pragma unroll
    for (int j = 0; j < 4; ++j) {
      v16bf vb = load_frag(lv, VSTR, j * 16, 0, lane);
      o[j] = wmma_bf16(pf.v, vb, o[j]);
    }
    __syncthreads();
  }

  const float inv = 1.f / ssum;
#pragma unroll
  for (int r = 0; r < 8; ++r) {
    const float ir = bcast(inv, half * 8 + r);
    const int n = qbase + half * 8 + r;
    const size_t row = (size_t)b * SEQ + n;
#pragma unroll
    for (int j = 0; j < 4; ++j)
      attn[row * DIM + h * DH + j * 16 + nsub] = f2bf(o[j][r] * ir);
  }
}

// ---------------- kernel: output projection (f32 result) ----------------
__global__ void __launch_bounds__(256) gemm_out_kernel(const ush* __restrict__ attn,
                                                       const ush* __restrict__ woutt,
                                                       float* __restrict__ out) {
  __shared__ ush lA[2 * LA_SZ];
  __shared__ ush lB[2 * LB_SZ];
  const int lane = threadIdx.x & 31, wave = threadIdx.x >> 5;
  const int m0 = blockIdx.x * 256, n0 = blockIdx.y * 64;
  v8f acc[2][4] = {};
  gemm_core(attn, woutt, m0, n0, wave, lane, lA, lB, acc);
  const int half = lane >> 4, nsub = lane & 15;
#pragma unroll
  for (int s = 0; s < 2; ++s)
#pragma unroll
    for (int j = 0; j < 4; ++j) {
      const int c = n0 + j * 16 + nsub;
#pragma unroll
      for (int r = 0; r < 8; ++r) {
        const int R = m0 + wave * 32 + s * 16 + half * 8 + r;
        out[(size_t)R * DIM + c] = acc[s][j][r];
      }
    }
}

// ---------------- launch ----------------
extern "C" void kernel_launch(void* const* d_in, const int* in_sizes, int n_in,
                              void* d_out, int out_size, void* d_ws, size_t ws_size,
                              hipStream_t stream) {
  const float*         x      = (const float*)d_in[0];
  const unsigned char* cmask  = (const unsigned char*)d_in[1];
  const float*         gamma  = (const float*)d_in[2];
  const float*         nullkv = (const float*)d_in[3];
  const float*         wq     = (const float*)d_in[4];
  const float*         wkv    = (const float*)d_in[5];
  const float*         wout   = (const float*)d_in[6];
  float* out = (float*)d_out;
  char*  ws  = (char*)d_ws;

  ush* xn    = (ush*)(ws + XN_OFF);     // also attn-out bf16
  ush* wqt   = (ush*)(ws + WQT_OFF);
  ush* wkvt  = (ush*)(ws + WKVT_OFF);
  ush* woutt = (ush*)(ws + WOUTT_OFF);
  ush* qbuf  = (ush*)(ws + Q_OFF);
  ush* kbuf  = (ush*)(ws + K_OFF);
  ush* vtbuf = (ush*)(ws + VT_OFF);

  castT_kernel<<<(DIM * DIM + 255) / 256, 256, 0, stream>>>(wq, wqt, DIM, DIM);
  castT_kernel<<<(DIM * 2048 + 255) / 256, 256, 0, stream>>>(wkv, wkvt, DIM, 2048);
  castT_kernel<<<(DIM * DIM + 255) / 256, 256, 0, stream>>>(wout, woutt, DIM, DIM);
  init_kv_kernel<<<BATCH * HEADS, DH, 0, stream>>>(nullkv, kbuf, vtbuf);
  ln_kernel<<<ROWS, 256, 0, stream>>>(x, gamma, xn);
  gemm_q_kernel<<<dim3(ROWS / 256, DIM / 64), 256, 0, stream>>>(xn, wqt, qbuf);
  gemm_kv_kernel<<<dim3(ROWS / 256, 2048 / 64), 256, 0, stream>>>(xn, wkvt, kbuf, vtbuf);
  attn_kernel<<<BATCH * HEADS * (SEQ / 128), 256, 0, stream>>>(qbuf, kbuf, vtbuf, cmask, xn);
  gemm_out_kernel<<<dim3(ROWS / 256, DIM / 64), 256, 0, stream>>>(xn, woutt, out);
}
